// PackableQuantLinear_5755256177094
// MI455X (gfx1250) — matrix operand order repaired and checked by reading, not compiled, not run on hardware
//
#include <hip/hip_runtime.h>

// GPTQ 4-bit dequant + GEMM for MI455X (gfx1250, wave32, WMMA).
// out[M,N] = x[M,K] @ dequant(qweight)[K,N] + bias,  dequant = scale*(q - zero)
// Double-buffered LDS pipeline: global loads for tile t+1 issue before the
// WMMAs of tile t, convert/dequant+LDS-store after, one barrier per tile.

typedef _Float16 v16h __attribute__((ext_vector_type(16)));
typedef _Float16 v8h  __attribute__((ext_vector_type(8)));
typedef _Float16 v4h  __attribute__((ext_vector_type(4)));
typedef float    v8f  __attribute__((ext_vector_type(8)));
typedef float    v4f  __attribute__((ext_vector_type(4)));

#define BM 128
#define BN 128
#define BK 64
#define PADH 8              // pad in halves (16B) to spread LDS banks
#define LDA (BK + PADH)     // 72 halves = 144 B row stride (16B aligned)
#define LDB (BK + PADH)
#define QGROUP 128          // quantization group size along K
#define A_ITERS ((BM * BK) / (256 * 4))   // 8 float4 loads per thread

union FragU { v16h v; v8h h[2]; };

__global__ __launch_bounds__(256)
void gptq_wmma_gemm(const float* __restrict__ x,
                    const int*   __restrict__ qweight,   // [K/8, N] packed nibbles
                    const int*   __restrict__ qzeros,    // [K/QGROUP, N/8]
                    const float* __restrict__ scales,    // [K/QGROUP, N]
                    const float* __restrict__ bias,      // [N]
                    float*       __restrict__ out,       // [M, N]
                    int M, int N, int K)
{
    // double-buffered tiles: 2*(18432 + 18432) = 73728 B (< 320 KB/WGP on CDNA5)
    __shared__ _Float16 sA[2][BM * LDA];
    __shared__ _Float16 sB[2][BN * LDB];

    const int tid  = threadIdx.x;
    const int lane = tid & 31;
    const int wv   = tid >> 5;     // wave 0..7
    const int wm   = wv & 1;       // 2 waves along M
    const int wn   = wv >> 1;      // 4 waves along N
    const int lm   = lane & 15;
    const int lh   = lane >> 4;    // half-wave select

    const int n0 = blockIdx.x * BN;
    const int m0 = blockIdx.y * BM;

    // B-staging assignment: each thread owns one column n and 4 qweight rows
    const int nl   = tid & 127;    // local n
    const int rsel = tid >> 7;     // 0/1 -> rows [0..3] or [4..7] of the k-tile
    const int nB   = n0 + nl;

    v8f acc[4][2];
#pragma unroll
    for (int i = 0; i < 4; ++i)
#pragma unroll
        for (int j = 0; j < 2; ++j)
#pragma unroll
            for (int e = 0; e < 8; ++e) acc[i][j][e] = 0.0f;

    // ---- pipeline stage helpers -------------------------------------------
    // Issue global loads for k-tile t into registers (no dependent math here,
    // so no waits are forced until the store phase consumes them).
    auto load_tile = [&](int t, v4f aR[A_ITERS], unsigned bR[4],
                         float& scR, unsigned& zwR) {
        const int k0 = t * BK;
        const float* xt = x + (size_t)m0 * K + k0;
#pragma unroll
        for (int it = 0; it < A_ITERS; ++it) {
            const int e = it * 1024 + tid * 4;
            aR[it] = *(const v4f*)(xt + (size_t)(e >> 6) * K + (e & (BK - 1)));
        }
        const int g = k0 / QGROUP;
        scR = scales[(size_t)g * N + nB];
        zwR = (unsigned)qzeros[(size_t)g * (N >> 3) + (nB >> 3)];
        const int* qrow = qweight + (size_t)((k0 >> 3) + rsel * 4) * N + nB;
#pragma unroll
        for (int i = 0; i < 4; ++i)
            bR[i] = (unsigned)qrow[(size_t)i * N];
    };

    // Convert / dequant the registers and store into LDS buffer `buf`.
    auto store_tile = [&](int buf, const v4f aR[A_ITERS], const unsigned bR[4],
                          float scR, unsigned zwR) {
        _Float16* A = &sA[buf][0];
        _Float16* B = &sB[buf][0];
#pragma unroll
        for (int it = 0; it < A_ITERS; ++it) {
            const int e = it * 1024 + tid * 4;
            v4h h;
            h[0] = (_Float16)aR[it][0]; h[1] = (_Float16)aR[it][1];
            h[2] = (_Float16)aR[it][2]; h[3] = (_Float16)aR[it][3];
            *(v4h*)&A[(e >> 6) * LDA + (e & (BK - 1))] = h;   // ds_store_b64
        }
        const float zs = scR * (float)((zwR >> ((nB & 7) * 4)) & 15u);
#pragma unroll
        for (int i = 0; i < 4; ++i) {
            const unsigned w = bR[i];
            v8h h;
#pragma unroll
            for (int j = 0; j < 8; ++j)
                h[j] = (_Float16)(scR * (float)((w >> (4 * j)) & 15u) - zs);
            *(v8h*)&B[nl * LDB + (rsel * 4 + i) * 8] = h;     // ds_store_b128
        }
    };

    // 16 WMMAs against LDS buffer `buf`.
    auto compute_tile = [&](int buf) {
        const _Float16* A = &sA[buf][0];
        const _Float16* B = &sB[buf][0];
#pragma unroll
        for (int kk = 0; kk < BK / 32; ++kk) {
            FragU af[4];
            FragU bf[2];
#pragma unroll
            for (int ti = 0; ti < 4; ++ti) {
                // A lane layout: VGPR0-3 hold K=8*lh+0..7, VGPR4-7 hold K=16+8*lh+0..7
                const _Float16* p = &A[(wm * 64 + ti * 16 + lm) * LDA + kk * 32 + lh * 8];
                af[ti].h[0] = *(const v8h*)p;          // ds_load_b128
                af[ti].h[1] = *(const v8h*)(p + 16);   // ds_load_b128
            }
#pragma unroll
            for (int tj = 0; tj < 2; ++tj) {
                // B lane layout: VGPR0-3 hold K=8*lh+0..7, VGPR4-7 hold K=16+8*lh+0..7 (n=lm)
                const _Float16* p = &B[(wn * 32 + tj * 16 + lm) * LDB + kk * 32 + lh * 8];
                bf[tj].h[0] = *(const v8h*)p;
                bf[tj].h[1] = *(const v8h*)(p + 16);
            }
#pragma unroll
            for (int ti = 0; ti < 4; ++ti)
#pragma unroll
                for (int tj = 0; tj < 2; ++tj)
                    acc[ti][tj] = __builtin_amdgcn_wmma_f32_16x16x32_f16(
                        false, af[ti].v, false, bf[tj].v,
                        (short)0, acc[ti][tj], false, false);
        }
    };
    // -----------------------------------------------------------------------

    const int T = K / BK;

    // Prologue: fill buffer 0.
    {
        v4f aR[A_ITERS]; unsigned bR[4]; float scR; unsigned zwR;
        load_tile(0, aR, bR, scR, zwR);
        store_tile(0, aR, bR, scR, zwR);
    }
    __syncthreads();

    for (int t = 0; t < T; ++t) {
        v4f aR[A_ITERS]; unsigned bR[4]; float scR; unsigned zwR;
        const bool more = (t + 1 < T);

        if (more) load_tile(t + 1, aR, bR, scR, zwR);   // issue early, no wait

        if (t + 2 < T) {                                 // stream t+2 into L2
            const int kp = (t + 2) * BK;
            __builtin_prefetch(x + (size_t)(m0 + ((tid * 4) >> 6)) * K + kp + ((tid * 4) & (BK - 1)), 0, 0);
            __builtin_prefetch(qweight + (size_t)((kp >> 3) + rsel * 4) * N + nB, 0, 0);
        }

        compute_tile(t & 1);                             // WMMAs hide the loads

        if (more) store_tile((t + 1) & 1, aR, bR, scR, zwR);

        __syncthreads();
    }

    // ---- epilogue: C/D layout is M = r + 8*lh, N = lm within each 16x16 tile
#pragma unroll
    for (int tj = 0; tj < 2; ++tj) {
        const int   n  = n0 + wn * 32 + tj * 16 + lm;
        const float bv = bias[n];
#pragma unroll
        for (int ti = 0; ti < 4; ++ti) {
            const int mb = m0 + wm * 64 + ti * 16 + lh * 8;
#pragma unroll
            for (int r = 0; r < 8; ++r)
                out[(size_t)(mb + r) * N + n] = acc[ti][tj][r] + bv;
        }
    }
}

extern "C" void kernel_launch(void* const* d_in, const int* in_sizes, int n_in,
                              void* d_out, int out_size, void* d_ws, size_t ws_size,
                              hipStream_t stream) {
    const float* x       = (const float*)d_in[0];
    const int*   qweight = (const int*)d_in[1];
    const int*   qzeros  = (const int*)d_in[2];
    const float* scales  = (const float*)d_in[3];
    const float* bias    = (const float*)d_in[4];
    // d_in[5] = g_idx: contiguous grouping (i / 128), folded into index math.

    const int in_f   = in_sizes[5];           // K  (4096)
    const int out_f  = in_sizes[4];           // N  (11008)
    const int tokens = in_sizes[0] / in_f;    // M  (8192)

    dim3 grid(out_f / BN, tokens / BM);       // (86, 64) for the reference shapes
    gptq_wmma_gemm<<<grid, 256, 0, stream>>>(x, qweight, qzeros, scales, bias,
                                             (float*)d_out, tokens, out_f, in_f);
}
